// MixLinear_GEMM_4501125726801
// MI455X (gfx1250) — 8-bit matrix operand order repaired and checked by reading, compiled, not run on hardware
//
#include <hip/hip_runtime.h>
#include <hip/hip_bf16.h>

typedef __attribute__((ext_vector_type(8))) int v8i;

__global__ void k_colmask(const float* __restrict__ x, const float* __restrict__ sigmap,
                          int M, int K, int* __restrict__ mask) {
    int k = blockIdx.x * blockDim.x + threadIdx.x;
    if (k >= K) return;
    float sg = *sigmap;
    int found = 0;
    for (int m = 0; m < M; ++m) {
        if (fabsf(x[(size_t)m * K + k]) > sg) { found = 1; break; }
    }
    mask[k] = found;
}

__global__ void k_build_list(const int* __restrict__ mask, int K,
                             int* __restrict__ list, int* __restrict__ counter) {
    if (threadIdx.x != 0 || blockIdx.x != 0) return;
    int c = 0;
    for (int k = 0; k < K; ++k)
        if (mask[k]) list[c++] = k;
    *counter = c;
}

__global__ void __launch_bounds__(256) k_quant_rows(
    const float* __restrict__ x, const int* __restrict__ mask, int K,
    signed char* __restrict__ qx, float* __restrict__ xscale) {
    __shared__ float red[256];
    const int m = blockIdx.x;
    const float* row = x + (size_t)m * K;

    float mx = 0.f;
    for (int k = threadIdx.x; k < K; k += 256) {
        float v = mask[k] ? 0.f : row[k];
        mx = fmaxf(mx, fabsf(v));
    }
    red[threadIdx.x] = mx;
    __syncthreads();
    for (int s = 128; s > 0; s >>= 1) {
        if (threadIdx.x < s) red[threadIdx.x] = fmaxf(red[threadIdx.x], red[threadIdx.x + s]);
        __syncthreads();
    }
    float scale = fmaxf(red[0] * (1.0f / 127.0f), 1e-8f);
    if (threadIdx.x == 0) xscale[m] = scale;

    int* q4 = (int*)(qx + (size_t)m * K);
    for (int k4 = threadIdx.x; k4 < (K >> 2); k4 += 256) {
        int packed = 0;
#pragma unroll
        for (int j = 0; j < 4; ++j) {
            int k = k4 * 4 + j;
            float v = mask[k] ? 0.f : row[k];
            float q = rintf(v / scale);
            q = fminf(fmaxf(q, -127.f), 127.f);
            packed |= ((int)q & 0xFF) << (8 * j);
        }
        q4[k4] = packed;
    }
}

__global__ void k_prep_w(const int* __restrict__ qw, const int* __restrict__ mask,
                         int K, long total4, int* __restrict__ qw8) {
    long i = (long)blockIdx.x * blockDim.x + threadIdx.x;
    if (i >= total4) return;
    long base = i * 4;
    int k = (int)(base % K);
    int packed = 0;
#pragma unroll
    for (int j = 0; j < 4; ++j) {
        int v = mask[k + j] ? 0 : qw[base + j];
        packed |= (v & 0xFF) << (8 * j);
    }
    qw8[i] = packed;
}

#define OCHUNK 64
__global__ void __launch_bounds__(256) k_init_out(
    const float* __restrict__ x, const int* __restrict__ qw,
    const float* __restrict__ scale_col, const float* __restrict__ bias,
    const int* __restrict__ list, const int* __restrict__ counter,
    int M, int N, int K, float* __restrict__ out) {
    __shared__ float xs[32][OCHUNK];
    __shared__ float wv[32][OCHUNK];
    const int m0 = blockIdx.y * 32, n0 = blockIdx.x * 32;
    const int tid = threadIdx.x;
    const int tn = tid & 31;
    const int tg = tid >> 5;
    const int cnt = *counter;

    float acc[4] = {0.f, 0.f, 0.f, 0.f};
    for (int c0 = 0; c0 < cnt; c0 += OCHUNK) {
        int cc = cnt - c0; if (cc > OCHUNK) cc = OCHUNK;
        for (int j = tid; j < 32 * OCHUNK; j += 256) {
            int r = j / OCHUNK, ci = j % OCHUNK;
            float xv = 0.f, wvv = 0.f;
            if (ci < cc) {
                int k = list[c0 + ci];
                xv  = x[(size_t)(m0 + r) * K + k];
                wvv = (float)qw[(size_t)(n0 + r) * K + k];
            }
            xs[r][ci] = xv; wv[r][ci] = wvv;
        }
        __syncthreads();
        for (int ci = 0; ci < cc; ++ci) {
            float w = wv[tn][ci];
#pragma unroll
            for (int r = 0; r < 4; ++r) acc[r] += xs[tg * 4 + r][ci] * w;
        }
        __syncthreads();
    }
    float sc = scale_col[n0 + tn], bv = bias[n0 + tn];
#pragma unroll
    for (int r = 0; r < 4; ++r) {
        int m = m0 + tg * 4 + r;
        out[(size_t)m * N + (n0 + tn)] = bv + sc * acc[r];
    }
}

__device__ __forceinline__ v8i load_a_frag(const signed char* p) {
    int2 p0 = *(const int2*)(p);
    int2 p1 = *(const int2*)(p + 8);
    int2 p2 = *(const int2*)(p + 32);
    int2 p3 = *(const int2*)(p + 40);
    v8i a;
    a[0] = p0.x; a[1] = p0.y; a[2] = p1.x; a[3] = p1.y;
    a[4] = p2.x; a[5] = p2.y; a[6] = p3.x; a[7] = p3.y;
    return a;
}
__device__ __forceinline__ v8i load_b_frag(const signed char* p) {
    int4 q0 = *(const int4*)(p);
    int4 q1 = *(const int4*)(p + 32);
    v8i b;
    b[0] = q0.x; b[1] = q0.y; b[2] = q0.z; b[3] = q0.w;
    b[4] = q1.x; b[5] = q1.y; b[6] = q1.z; b[7] = q1.w;
    return b;
}

__global__ void __launch_bounds__(256) k_gemm_i8(
    const signed char* __restrict__ qx,
    const signed char* __restrict__ qw8,
    const float* __restrict__ xscale,
    const float* __restrict__ scale_col,
    float* __restrict__ out,
    int M, int N, int K) {
    const int lane   = threadIdx.x & 31;
    const int wave   = threadIdx.x >> 5;
    const int wm     = wave >> 1;
    const int wn     = wave & 1;
    const int m0     = blockIdx.y * 128 + wm * 32;
    const int n0     = blockIdx.x * 128 + wn * 64;
    const int lane16 = lane & 15;
    const int laneHi = lane >> 4;

    const signed char* ap[2];
    ap[0] = qx + (size_t)(m0 + lane16) * K + laneHi * 16;
    ap[1] = ap[0] + (size_t)16 * K;
    const signed char* bp[4];
    bp[0] = qw8 + (size_t)(n0 + lane16) * K + laneHi * 16;
    bp[1] = bp[0] + (size_t)16 * K;
    bp[2] = bp[0] + (size_t)32 * K;
    bp[3] = bp[0] + (size_t)48 * K;

    v8i acc[2][4];
#pragma unroll
    for (int i = 0; i < 2; ++i)
#pragma unroll
        for (int j = 0; j < 4; ++j) acc[i][j] = {};

    for (int k = 0; k < K; k += 64) {
        v8i a[2], b[4];
#pragma unroll
        for (int i = 0; i < 2; ++i) a[i] = load_a_frag(ap[i] + k);
#pragma unroll
        for (int j = 0; j < 4; ++j) b[j] = load_b_frag(bp[j] + k);
#pragma unroll
        for (int i = 0; i < 2; ++i)
#pragma unroll
            for (int j = 0; j < 4; ++j)
                acc[i][j] = __builtin_amdgcn_wmma_i32_16x16x64_iu8(
                    true, a[i], true, b[j], acc[i][j], false, false);
    }

#pragma unroll
    for (int i = 0; i < 2; ++i) {
        const int mrow = m0 + i * 16 + laneHi * 8;
        float xs8[8];
#pragma unroll
        for (int r = 0; r < 8; ++r) xs8[r] = xscale[mrow + r];
#pragma unroll
        for (int j = 0; j < 4; ++j) {
            const int ncol = n0 + j * 16 + lane16;
            const float sc = scale_col[ncol];
#pragma unroll
            for (int r = 0; r < 8; ++r) {
                float* p = out + (size_t)(mrow + r) * N + ncol;
                *p += (float)acc[i][j][r] * xs8[r] * sc;
            }
        }
    }
}

extern "C" void kernel_launch(void* const* d_in, const int* in_sizes, int n_in,
                              void* d_out, int out_size, void* d_ws, size_t ws_size,
                              hipStream_t stream) {
    const float* x         = (const float*)d_in[0];
    const int*   qw        = (const int*)d_in[1];
    const float* scale_col = (const float*)d_in[2];
    const float* sigmap    = (const float*)d_in[3];
    const float* bias      = (const float*)d_in[4];
    float* out = (float*)d_out;

    const int N = in_sizes[2];
    const int K = in_sizes[1] / N;
    const int M = in_sizes[0] / K;

    char* ws = (char*)d_ws;
    size_t off = 0;
    signed char* qx  = (signed char*)(ws + off); off += (size_t)M * K;
    signed char* qw8 = (signed char*)(ws + off); off += (size_t)N * K;
    float* xscale    = (float*)(ws + off);       off += (size_t)M * sizeof(float);
    int*   mask      = (int*)(ws + off);         off += (size_t)K * sizeof(int);
    int*   list      = (int*)(ws + off);         off += (size_t)K * sizeof(int);
    int*   counter   = (int*)(ws + off);         off += sizeof(int);

    k_colmask<<<(K + 255) / 256, 256, 0, stream>>>(x, sigmap, M, K, mask);
    k_build_list<<<1, 1, 0, stream>>>(mask, K, list, counter);
    k_quant_rows<<<M, 256, 0, stream>>>(x, mask, K, qx, xscale);

    long total4 = (long)N * K / 4;
    k_prep_w<<<(unsigned)((total4 + 255) / 256), 256, 0, stream>>>(qw, mask, K, total4, (int*)qw8);

    dim3 gInit(N / 32, M / 32);
    k_init_out<<<gInit, 256, 0, stream>>>(x, qw, scale_col, bias, list, counter, M, N, K, out);

    dim3 gGemm(N / 128, M / 128);
    k_gemm_i8<<<gGemm, 256, 0, stream>>>(qx, qw8, xscale, scale_col, out, M, N, K);
}
